// CausalVulGNN_81758997447205
// MI455X (gfx1250) — matrix operand order repaired and checked by reading, not compiled, not run on hardware
//
#include <hip/hip_runtime.h>
#include <hip/hip_bf16.h>

// ---------------------------------------------------------------------------
// Problem constants (from the reference)
// ---------------------------------------------------------------------------
#define NN   50000
#define EE   800000
#define GG   64
#define RR   5
#define BERT 768
#define HH   128
#define DD   32
#define CC   2
#define EPS_BN  1e-5f
#define BETA_BN 1e-4f

typedef __attribute__((ext_vector_type(16))) __bf16         v16bf;
typedef __attribute__((ext_vector_type(8)))  float          v8f;
typedef __attribute__((ext_vector_type(8)))  unsigned short v8us;

union FragBF { v16bf v; unsigned short u[16]; v8us h[2]; };

__device__ __forceinline__ unsigned short f2bf(float f) {
  // round-to-nearest-even fp32 -> bf16 bit pattern
  unsigned int x = __float_as_uint(f);
  unsigned int r = (x + 0x7fffu + ((x >> 16) & 1u)) >> 16;
  return (unsigned short)r;
}

// ---------------------------------------------------------------------------
// WMMA GEMM:  C[M,Nc] = A[M,K](bf16) @ B (as Bt[Nc,K] bf16) + bias
// One wave computes a 16x32 tile: one A fragment feeds two
// v_wmma_f32_16x16x32_bf16 issues (register blocking on N).
// Requires: M%16==0 (whole-wave guard for grid overshoot), K%32==0, Nc%32==0.
// A-fragment (16-bit A 16x32): lane l<16 -> row l, elems 0..15 = K k0..k0+15,
// lanes 16..31 take K k0+16..k0+31 (same contiguous order as B).
// B-fragment from Bt (row = output column): elems i -> K = k0 + i + (lane/16)*16.
// C/D: VGPR j -> row m0 + j + (lane/16)*8, col n0 + (lane&15).
// ---------------------------------------------------------------------------
template <int RELU>
__global__ void gemm_wmma_bf16(const unsigned short* __restrict__ A,
                               const unsigned short* __restrict__ Bt,
                               const float* __restrict__ bias,
                               float* __restrict__ C,
                               int M, int K, int Nc) {
  const int lane  = threadIdx.x & 31;
  const int wave  = threadIdx.x >> 5;
  const int mtile = blockIdx.x * 4 + wave;
  if (mtile * 16 >= M) return;                 // uniform per-wave exit (EXEC stays all-1)
  const int m0  = mtile * 16;
  const int n0  = blockIdx.y * 32;
  const int l15 = lane & 15;
  const int hi  = lane >> 4;

  const unsigned short* Arow  = A  + (size_t)(m0 + l15) * K;
  const unsigned short* Brow0 = Bt + (size_t)(n0 + l15) * K;
  const unsigned short* Brow1 = Bt + (size_t)(n0 + 16 + l15) * K;

  v8f acc0 = {};
  v8f acc1 = {};
  for (int k0 = 0; k0 < K; k0 += 32) {
    FragBF a, b0, b1;
    a.h[0]  = *(const v8us*)(Arow + k0 + hi * 16);       // K k0..k0+7 (+16 for hi lanes)
    a.h[1]  = *(const v8us*)(Arow + k0 + hi * 16 + 8);   // K k0+8..   (+16 for hi lanes)
    b0.h[0] = *(const v8us*)(Brow0 + k0 + hi * 16);
    b0.h[1] = *(const v8us*)(Brow0 + k0 + hi * 16 + 8);
    b1.h[0] = *(const v8us*)(Brow1 + k0 + hi * 16);
    b1.h[1] = *(const v8us*)(Brow1 + k0 + hi * 16 + 8);
    __builtin_prefetch(Arow + k0 + 32, 0, 0);            // speculative next-chunk prefetch

    acc0 = __builtin_amdgcn_wmma_f32_16x16x32_bf16(
        false, a.v, false, b0.v, (short)0, acc0, false, false);
    acc1 = __builtin_amdgcn_wmma_f32_16x16x32_bf16(
        false, a.v, false, b1.v, (short)0, acc1, false, false);
  }

  const int   c0  = n0 + l15;
  const int   c1  = c0 + 16;
  const float bv0 = bias[c0];
  const float bv1 = bias[c1];
#pragma unroll
  for (int j = 0; j < 8; ++j) {
    float v0 = acc0[j] + bv0;
    float v1 = acc1[j] + bv1;
    if (RELU) { v0 = fmaxf(v0, 0.0f); v1 = fmaxf(v1, 0.0f); }
    const size_t row = (size_t)(m0 + j + hi * 8);
    C[row * Nc + c0] = v0;
    C[row * Nc + c1] = v1;
  }
}

// ---------------------------------------------------------------------------
// Utility / graph kernels
// ---------------------------------------------------------------------------
__global__ void fill_zero(float* __restrict__ p, size_t n) {
  size_t i = (size_t)blockIdx.x * blockDim.x + threadIdx.x;
  size_t s = (size_t)gridDim.x * blockDim.x;
  for (; i < n; i += s) p[i] = 0.0f;
}

__global__ void convert_f32_bf16(const float* __restrict__ X,
                                 unsigned short* __restrict__ Y, size_t n) {
  size_t i = (size_t)blockIdx.x * blockDim.x + threadIdx.x;
  size_t s = (size_t)gridDim.x * blockDim.x;
  for (; i < n; i += s) Y[i] = f2bf(X[i]);
}

__global__ void transpose_to_bf16(const float* __restrict__ W,
                                  unsigned short* __restrict__ Bt,
                                  int K, int Ncols) {
  int idx = blockIdx.x * blockDim.x + threadIdx.x;
  if (idx >= K * Ncols) return;
  int n = idx / K, k = idx % K;               // Bt[n*K + k] = W[k, n]
  Bt[idx] = f2bf(W[(size_t)k * Ncols + n]);
}

__global__ void degrees_k(const int* __restrict__ src, const int* __restrict__ dst,
                          float* __restrict__ dout, float* __restrict__ din, int E) {
  int e = blockIdx.x * blockDim.x + threadIdx.x;
  if (e >= E) return;
  atomicAdd(&dout[src[e]], 1.0f);
  atomicAdd(&din[dst[e]], 1.0f);
}

__global__ void finalize_norms(const float* __restrict__ dout, const float* __restrict__ din,
                               float* __restrict__ norm_s, float* __restrict__ norm_d, int N) {
  int n = blockIdx.x * blockDim.x + threadIdx.x;
  if (n >= N) return;
  norm_s[n] = rsqrtf(fmaxf(dout[n], 1.0f));
  norm_d[n] = rsqrtf(fmaxf(din[n], 1.0f));
}

// BN statistics over rows for C=128 columns (LDS-staged atomics -> global)
__global__ void bn_stats128(const float* __restrict__ X, float* __restrict__ sums,
                            float* __restrict__ sumsq, int Nrows) {
  __shared__ float ss[128];
  __shared__ float sq[128];
  if (threadIdx.x < 128) { ss[threadIdx.x] = 0.0f; sq[threadIdx.x] = 0.0f; }
  __syncthreads();
  size_t total = (size_t)Nrows * 128;
  for (size_t i = (size_t)blockIdx.x * blockDim.x + threadIdx.x; i < total;
       i += (size_t)gridDim.x * blockDim.x) {
    float v = X[i];
    int   c = (int)(i & 127);
    atomicAdd(&ss[c], v);
    atomicAdd(&sq[c], v * v);
  }
  __syncthreads();
  if (threadIdx.x < 128) {
    atomicAdd(&sums[threadIdx.x], ss[threadIdx.x]);
    atomicAdd(&sumsq[threadIdx.x], sq[threadIdx.x]);
  }
}

__global__ void bn_finalize128(float* __restrict__ sums, float* __restrict__ sumsq,
                               float nrows) {
  int c = threadIdx.x;
  if (c < 128) {
    float m = sums[c] / nrows;
    float v = sumsq[c] / nrows - m * m;
    sums[c]  = m;
    sumsq[c] = rsqrtf(v + EPS_BN);
  }
}

// Y(f32) = ((X - mean)*rstd + beta) * rowscale[n]   (message-path producer)
__global__ void bn_apply128_f32(const float* __restrict__ X, const float* __restrict__ mean,
                                const float* __restrict__ rstd,
                                const float* __restrict__ rowscale,
                                float* __restrict__ Y, int Nrows) {
  size_t i = (size_t)blockIdx.x * blockDim.x + threadIdx.x;
  if (i >= (size_t)Nrows * 128) return;
  int    c = (int)(i & 127);
  size_t n = i >> 7;
  Y[i] = ((X[i] - mean[c]) * rstd[c] + BETA_BN) * rowscale[n];
}

// Y(bf16) = (X - mean)*rstd + beta   (GEMM A-operand producer)
__global__ void bn_apply128_bf16(const float* __restrict__ X, const float* __restrict__ mean,
                                 const float* __restrict__ rstd,
                                 unsigned short* __restrict__ Y, int Nrows) {
  size_t i = (size_t)blockIdx.x * blockDim.x + threadIdx.x;
  if (i >= (size_t)Nrows * 128) return;
  int c = (int)(i & 127);
  Y[i] = f2bf((X[i] - mean[c]) * rstd[c] + BETA_BN);
}

// RelGraphConv message scatter: agg[dst] += xproj[src][etype*128 + h]
__global__ void rel_scatter(const float* __restrict__ xproj, const int* __restrict__ src,
                            const int* __restrict__ dst, const int* __restrict__ etype,
                            float* __restrict__ agg, int E) {
  int idx = blockIdx.x * blockDim.x + threadIdx.x;
  if (idx >= E * 128) return;
  int e = idx >> 7, h = idx & 127;
  float m = xproj[(size_t)src[e] * (RR * 128) + etype[e] * 128 + h];
  atomicAdd(&agg[(size_t)dst[e] * 128 + h], m);
}

// GraphConv message scatter with optional edge weight column from [E,2]
__global__ void edge_scatter(const float* __restrict__ tmp, const int* __restrict__ src,
                             const int* __restrict__ dst, const float* __restrict__ ew2,
                             int col, float* __restrict__ agg, int E) {
  int idx = blockIdx.x * blockDim.x + threadIdx.x;
  if (idx >= E * 128) return;
  int e = idx >> 7, h = idx & 127;
  float m = tmp[(size_t)src[e] * 128 + h];
  if (ew2) m *= ew2[(size_t)e * 2 + col];
  atomicAdd(&agg[(size_t)dst[e] * 128 + h], m);
}

// Ybf16[n,h] = agg[n,h] * norm_d[n]   (aggregate -> GEMM A-operand)
__global__ void scale_rows_bf16(const float* __restrict__ X, const float* __restrict__ norm_d,
                                unsigned short* __restrict__ Y, int Nrows) {
  size_t i = (size_t)blockIdx.x * blockDim.x + threadIdx.x;
  if (i >= (size_t)Nrows * 128) return;
  Y[i] = f2bf(X[i] * norm_d[i >> 7]);
}

__global__ void relu_inplace(float* __restrict__ X, size_t n) {
  size_t i = (size_t)blockIdx.x * blockDim.x + threadIdx.x;
  if (i < n) X[i] = fmaxf(X[i], 0.0f);
}

__global__ void node_attention_k(const float* __restrict__ x, const float* __restrict__ W,
                                 const float* __restrict__ b, float* __restrict__ natt, int N) {
  int n = blockIdx.x * blockDim.x + threadIdx.x;
  if (n >= N) return;
  const float* xr = x + (size_t)n * 128;
  float s0 = b[0], s1 = b[1];
#pragma unroll 4
  for (int h = 0; h < 128; ++h) {
    float v = xr[h];
    s0 += v * W[h * 2 + 0];
    s1 += v * W[h * 2 + 1];
  }
  float m = fmaxf(s0, s1);
  float e0 = __expf(s0 - m), e1 = __expf(s1 - m);
  float inv = 1.0f / (e0 + e1);
  natt[(size_t)n * 2 + 0] = e0 * inv;
  natt[(size_t)n * 2 + 1] = e1 * inv;
}

__global__ void edge_attention_k(const float* __restrict__ x, const int* __restrict__ src,
                                 const int* __restrict__ dst, const float* __restrict__ W,
                                 const float* __restrict__ b, float* __restrict__ eatt, int E) {
  __shared__ float w[512];
  __shared__ float bb[2];
  for (int i = threadIdx.x; i < 512; i += blockDim.x) w[i] = W[i];
  if (threadIdx.x < 2) bb[threadIdx.x] = b[threadIdx.x];
  __syncthreads();
  int e = blockIdx.x * blockDim.x + threadIdx.x;
  if (e >= E) return;
  const float* xs = x + (size_t)src[e] * 128;
  const float* xd = x + (size_t)dst[e] * 128;
  float s0 = bb[0], s1 = bb[1];
#pragma unroll 4
  for (int h = 0; h < 128; ++h) {
    float vs = xs[h], vd = xd[h];
    s0 += vs * w[h * 2 + 0] + vd * w[(128 + h) * 2 + 0];
    s1 += vs * w[h * 2 + 1] + vd * w[(128 + h) * 2 + 1];
  }
  float m = fmaxf(s0, s1);
  float e0 = __expf(s0 - m), e1 = __expf(s1 - m);
  float inv = 1.0f / (e0 + e1);
  eatt[(size_t)e * 2 + 0] = e0 * inv;
  eatt[(size_t)e * 2 + 1] = e1 * inv;
}

__global__ void scale_by_att(const float* __restrict__ x, const float* __restrict__ natt,
                             float* __restrict__ xc, float* __restrict__ xs, int N) {
  size_t i = (size_t)blockIdx.x * blockDim.x + threadIdx.x;
  if (i >= (size_t)N * 128) return;
  size_t n = i >> 7;
  float  v = x[i];
  xc[i] = natt[n * 2 + 0] * v;
  xs[i] = natt[n * 2 + 1] * v;
}

__global__ void graph_count(const int* __restrict__ gid, float* __restrict__ cnt, int N) {
  int n = blockIdx.x * blockDim.x + threadIdx.x;
  if (n >= N) return;
  atomicAdd(&cnt[gid[n]], 1.0f);
}

__global__ void graph_pool_sum(const float* __restrict__ xn, const int* __restrict__ gid,
                               float* __restrict__ xg, int N) {
  size_t i = (size_t)blockIdx.x * blockDim.x + threadIdx.x;
  if (i >= (size_t)N * 128) return;
  size_t n = i >> 7;
  int    h = (int)(i & 127);
  atomicAdd(&xg[(size_t)gid[n] * 128 + h], xn[i]);
}

__global__ void graph_pool_div(float* __restrict__ xg, const float* __restrict__ cnt, int G) {
  int i = blockIdx.x * blockDim.x + threadIdx.x;
  if (i >= G * 128) return;
  xg[i] /= fmaxf(cnt[i >> 7], 1.0f);
}

// Small BN over rows (one block, cols threads)
__global__ void bn_small(const float* __restrict__ X, float* __restrict__ Y,
                         int rows, int cols) {
  int c = threadIdx.x;
  if (c >= cols) return;
  float s = 0.0f, q = 0.0f;
  for (int r = 0; r < rows; ++r) {
    float v = X[(size_t)r * cols + c];
    s += v;
    q += v * v;
  }
  float m   = s / rows;
  float var = q / rows - m * m;
  float rs  = rsqrtf(var + EPS_BN);
  for (int r = 0; r < rows; ++r)
    Y[(size_t)r * cols + c] = (X[(size_t)r * cols + c] - m) * rs + BETA_BN;
}

__global__ void mm_small(const float* __restrict__ X, const float* __restrict__ W,
                         const float* __restrict__ b, float* __restrict__ Y,
                         int rows, int K, int Nc, int relu) {
  int idx = blockIdx.x * blockDim.x + threadIdx.x;
  if (idx >= rows * Nc) return;
  int r = idx / Nc, n = idx % Nc;
  float s = b ? b[n] : 0.0f;
  for (int k = 0; k < K; ++k) s += X[(size_t)r * K + k] * W[(size_t)k * Nc + n];
  if (relu) s = fmaxf(s, 0.0f);
  Y[idx] = s;
}

__global__ void log_softmax2(float* __restrict__ Y, int rows) {
  int r = blockIdx.x * blockDim.x + threadIdx.x;
  if (r >= rows) return;
  float a = Y[r * 2], b = Y[r * 2 + 1];
  float m = fmaxf(a, b);
  float l = m + __logf(__expf(a - m) + __expf(b - m));
  Y[r * 2]     = a - l;
  Y[r * 2 + 1] = b - l;
}

// Deterministic permutation stand-in (bijection on 0..63; JAX PRNG not reproducible here)
__global__ void build_xcat(const float* __restrict__ xcg, const float* __restrict__ xsg,
                           float* __restrict__ xcat, int G) {
  int idx = blockIdx.x * blockDim.x + threadIdx.x;
  if (idx >= G * 256) return;
  int g = idx >> 8, c = idx & 255;
  if (c < 128) xcat[idx] = xcg[(size_t)g * 128 + c];
  else         xcat[idx] = xsg[(size_t)((37 * g + 11) & 63) * 128 + (c - 128)];
}

// ---------------------------------------------------------------------------
// Host orchestration
// ---------------------------------------------------------------------------
static inline int cdiv(long long a, long long b) { return (int)((a + b - 1) / b); }

extern "C" void kernel_launch(void* const* d_in, const int* in_sizes, int n_in,
                              void* d_out, int out_size, void* d_ws, size_t ws_size,
                              hipStream_t stream) {
  (void)in_sizes; (void)n_in; (void)out_size;
  const float* features = (const float*)d_in[0];
  const int*   src      = (const int*)d_in[1];
  const int*   dst      = (const int*)d_in[2];
  const int*   etype    = (const int*)d_in[3];
  const int*   gid      = (const int*)d_in[4];
  const float* W_red    = (const float*)d_in[5];
  const float* b_red    = (const float*)d_in[6];
  const float* W_rel    = (const float*)d_in[7];
  const float* W_self   = (const float*)d_in[8];
  const float* b_rel    = (const float*)d_in[9];
  const float* W_conv   = (const float*)d_in[10];
  const float* b_conv   = (const float*)d_in[11];
  const float* W_eatt   = (const float*)d_in[12];
  const float* b_eatt   = (const float*)d_in[13];
  const float* W_natt   = (const float*)d_in[14];
  const float* b_natt   = (const float*)d_in[15];
  const float* W_cconv  = (const float*)d_in[16];
  const float* b_cconv  = (const float*)d_in[17];
  const float* W_sconv  = (const float*)d_in[18];
  const float* b_sconv  = (const float*)d_in[19];
  const float* cfc1_W   = (const float*)d_in[20];
  const float* cfc1_b   = (const float*)d_in[21];
  const float* cfc2_W   = (const float*)d_in[22];
  const float* cfc2_b   = (const float*)d_in[23];
  const float* sfc1_W   = (const float*)d_in[24];
  const float* sfc1_b   = (const float*)d_in[25];
  const float* sfc2_W   = (const float*)d_in[26];
  const float* sfc2_b   = (const float*)d_in[27];
  const float* ccat1_W  = (const float*)d_in[28];
  const float* ccat1_b  = (const float*)d_in[29];
  const float* ccat2_W  = (const float*)d_in[30];
  const float* ccat2_b  = (const float*)d_in[31];

  float* out = (float*)d_out;
  float* out_causal = out;                   // [64,2]
  float* out_spur   = out + 128;             // [64,2]
  float* out_ctx    = out + 256;             // [64,2]
  float* xcg        = out + 384;             // [64,128]
  float* xsg        = out + 384 + 8192;      // [64,128]
  float* natt       = out + 384 + 16384;     // [N,2]
  float* eatt       = natt + (size_t)NN * 2; // [E,2]

  // ---- workspace bump allocator ----
  char*  w     = (char*)d_ws;
  size_t used  = 0;
  auto   alloc = [&](size_t bytes) -> char* {
    char* p = w + used;
    used += (bytes + 255) & ~(size_t)255;
    return p;
  };
  const size_t NHf = (size_t)NN * HH;                 // floats per node-feature buffer
  float* B0 = (float*)alloc(NHf * 4);
  float* B1 = (float*)alloc(NHf * 4);
  float* B2 = (float*)alloc(NHf * 4);
  float* B3 = (float*)alloc(NHf * 4);
  float* B4 = (float*)alloc(NHf * 4);
  float* B5 = (float*)alloc(NHf * 4);
  float* XP = (float*)alloc((size_t)NN * RR * HH * 4);  // x_proj [N, 5*128]
  unsigned short* XBF = (unsigned short*)alloc(NHf * 2);  // bn(x) as bf16 GEMM operand
  unsigned short* ABF = (unsigned short*)alloc(NHf * 2);  // scaled aggregate as bf16
  unsigned short* Wred_t  = (unsigned short*)alloc((size_t)HH * BERT * 2);
  unsigned short* Wrel_t  = (unsigned short*)alloc((size_t)RR * HH * HH * 2);
  unsigned short* Wself_t = (unsigned short*)alloc((size_t)HH * HH * 2);
  unsigned short* Wc0_t   = (unsigned short*)alloc((size_t)HH * HH * 2);
  unsigned short* Wc1_t   = (unsigned short*)alloc((size_t)HH * HH * 2);
  unsigned short* Wcc_t   = (unsigned short*)alloc((size_t)HH * HH * 2);
  unsigned short* Wsc_t   = (unsigned short*)alloc((size_t)HH * HH * 2);
  float* doutg  = (float*)alloc((size_t)NN * 4);
  float* ding   = (float*)alloc((size_t)NN * 4);
  float* norm_s = (float*)alloc((size_t)NN * 4);
  float* norm_d = (float*)alloc((size_t)NN * 4);
  float* sums   = (float*)alloc(128 * 4);
  float* sumsq  = (float*)alloc(128 * 4);
  float* cnt    = (float*)alloc(GG * 4);
  float* zbias  = (float*)alloc((size_t)RR * HH * 4);   // zero bias for x_proj GEMM
  float* t1     = (float*)alloc((size_t)GG * 256 * 4);  // head temp (up to [64,256])
  float* t2     = (float*)alloc((size_t)GG * DD * 4);
  float* t3     = (float*)alloc((size_t)GG * DD * 4);
  float* xcat   = (float*)alloc((size_t)GG * 256 * 4);
  float* xcatbn = (float*)alloc((size_t)GG * 256 * 4);
  if (used > ws_size) return;  // workspace too small: no-op (deterministic)

  // bf16 feature staging aliases the x_proj region (x_proj is written later)
  unsigned short* FBF = (unsigned short*)XP;   // N*768*2 = 76.8MB <= N*640*4

  const int BT = 256;
  const dim3 gBlk(128);                                    // 4 waves per block
  const int  mBlocks = cdiv(NN / 16, 4);                   // 50000/16 = 3125 tiles

  auto zero = [&](float* p, size_t n) {
    int blocks = cdiv((long long)n, BT);
    if (blocks > 65535) blocks = 65535;
    fill_zero<<<blocks, BT, 0, stream>>>(p, n);
  };
  auto bn_stats = [&](const float* X) {
    zero(sums, 128); zero(sumsq, 128);
    bn_stats128<<<512, BT, 0, stream>>>(X, sums, sumsq, NN);
    bn_finalize128<<<1, 128, 0, stream>>>(sums, sumsq, (float)NN);
  };

  // ---- weight transposes (fp32 -> bf16, [K,Nc] -> [Nc,K]) ----
  transpose_to_bf16<<<cdiv(BERT * HH, BT), BT, 0, stream>>>(W_red, Wred_t, BERT, HH);
  for (int r = 0; r < RR; ++r)
    transpose_to_bf16<<<cdiv(HH * HH, BT), BT, 0, stream>>>(W_rel + (size_t)r * HH * HH,
                                                            Wrel_t + (size_t)r * HH * HH, HH, HH);
  transpose_to_bf16<<<cdiv(HH * HH, BT), BT, 0, stream>>>(W_self, Wself_t, HH, HH);
  transpose_to_bf16<<<cdiv(HH * HH, BT), BT, 0, stream>>>(W_conv, Wc0_t, HH, HH);
  transpose_to_bf16<<<cdiv(HH * HH, BT), BT, 0, stream>>>(W_conv + HH * HH, Wc1_t, HH, HH);
  transpose_to_bf16<<<cdiv(HH * HH, BT), BT, 0, stream>>>(W_cconv, Wcc_t, HH, HH);
  transpose_to_bf16<<<cdiv(HH * HH, BT), BT, 0, stream>>>(W_sconv, Wsc_t, HH, HH);
  zero(zbias, (size_t)RR * HH);

  // ---- degrees & symmetric norms ----
  zero(doutg, NN); zero(ding, NN);
  degrees_k<<<cdiv(EE, BT), BT, 0, stream>>>(src, dst, doutg, ding, EE);
  finalize_norms<<<cdiv(NN, BT), BT, 0, stream>>>(doutg, ding, norm_s, norm_d, NN);

  // ---- x0 = features @ W_red + b_red  -> B0 ----
  convert_f32_bf16<<<4096, BT, 0, stream>>>(features, FBF, (size_t)NN * BERT);
  gemm_wmma_bf16<0><<<dim3(mBlocks, HH / 32), gBlk, 0, stream>>>(FBF, Wred_t, b_red, B0,
                                                                 NN, BERT, HH);
  // ---- BN -> XBF (bf16 operand) ----
  bn_stats(B0);
  bn_apply128_bf16<<<cdiv((long long)NHf, BT), BT, 0, stream>>>(B0, sums, sumsq, XBF, NN);

  // ---- RelGraphConv ----
  // x_proj (all 5 relations as one Nc=640 GEMM) -> XP (overwrites FBF alias, now dead)
  gemm_wmma_bf16<0><<<dim3(mBlocks, RR * HH / 32), gBlk, 0, stream>>>(XBF, Wrel_t, zbias, XP,
                                                                      NN, HH, RR * HH);
  // self-loop term into B1, then scatter messages, then relu. x := B1
  gemm_wmma_bf16<0><<<dim3(mBlocks, HH / 32), gBlk, 0, stream>>>(XBF, Wself_t, b_rel, B1,
                                                                 NN, HH, HH);
  rel_scatter<<<cdiv((long long)EE * HH, BT), BT, 0, stream>>>(XP, src, dst, etype, B1, EE);
  relu_inplace<<<cdiv((long long)NHf, BT), BT, 0, stream>>>(B1, NHf);

  // ---- two GraphConv layers ----
  // layer 0: x=B1 -> tmp=B2 -> agg=B0 -> out=B3
  bn_stats(B1);
  bn_apply128_f32<<<cdiv((long long)NHf, BT), BT, 0, stream>>>(B1, sums, sumsq, norm_s, B2, NN);
  zero(B0, NHf);
  edge_scatter<<<cdiv((long long)EE * HH, BT), BT, 0, stream>>>(B2, src, dst, nullptr, 0, B0, EE);
  scale_rows_bf16<<<cdiv((long long)NHf, BT), BT, 0, stream>>>(B0, norm_d, ABF, NN);
  gemm_wmma_bf16<1><<<dim3(mBlocks, HH / 32), gBlk, 0, stream>>>(ABF, Wc0_t, b_conv, B3,
                                                                 NN, HH, HH);
  // layer 1: x=B3 -> tmp=B2 -> agg=B1 -> out=B0
  bn_stats(B3);
  bn_apply128_f32<<<cdiv((long long)NHf, BT), BT, 0, stream>>>(B3, sums, sumsq, norm_s, B2, NN);
  zero(B1, NHf);
  edge_scatter<<<cdiv((long long)EE * HH, BT), BT, 0, stream>>>(B2, src, dst, nullptr, 0, B1, EE);
  scale_rows_bf16<<<cdiv((long long)NHf, BT), BT, 0, stream>>>(B1, norm_d, ABF, NN);
  gemm_wmma_bf16<1><<<dim3(mBlocks, HH / 32), gBlk, 0, stream>>>(ABF, Wc1_t, b_conv + HH, B0,
                                                                 NN, HH, HH);
  // x := B0

  // ---- attentions (written straight into d_out) ----
  node_attention_k<<<cdiv(NN, BT), BT, 0, stream>>>(B0, W_natt, b_natt, natt, NN);
  edge_attention_k<<<cdiv(EE, BT), BT, 0, stream>>>(B0, src, dst, W_eatt, b_eatt, eatt, EE);

  // ---- split causal/spurious ----
  scale_by_att<<<cdiv((long long)NHf, BT), BT, 0, stream>>>(B0, natt, B4, B5, NN);

  // causal: xc=B4 -> tmp=B2 -> agg=B1 -> x_causal=B3
  bn_stats(B4);
  bn_apply128_f32<<<cdiv((long long)NHf, BT), BT, 0, stream>>>(B4, sums, sumsq, norm_s, B2, NN);
  zero(B1, NHf);
  edge_scatter<<<cdiv((long long)EE * HH, BT), BT, 0, stream>>>(B2, src, dst, eatt, 0, B1, EE);
  scale_rows_bf16<<<cdiv((long long)NHf, BT), BT, 0, stream>>>(B1, norm_d, ABF, NN);
  gemm_wmma_bf16<1><<<dim3(mBlocks, HH / 32), gBlk, 0, stream>>>(ABF, Wcc_t, b_cconv, B3,
                                                                 NN, HH, HH);
  // spurious: xs=B5 -> tmp=B2 -> agg=B0 -> x_spurious=B4
  bn_stats(B5);
  bn_apply128_f32<<<cdiv((long long)NHf, BT), BT, 0, stream>>>(B5, sums, sumsq, norm_s, B2, NN);
  zero(B0, NHf);
  edge_scatter<<<cdiv((long long)EE * HH, BT), BT, 0, stream>>>(B2, src, dst, eatt, 1, B0, EE);
  scale_rows_bf16<<<cdiv((long long)NHf, BT), BT, 0, stream>>>(B0, norm_d, ABF, NN);
  gemm_wmma_bf16<1><<<dim3(mBlocks, HH / 32), gBlk, 0, stream>>>(ABF, Wsc_t, b_sconv, B4,
                                                                 NN, HH, HH);

  // ---- graph mean pooling into d_out ----
  zero(cnt, GG);
  graph_count<<<cdiv(NN, BT), BT, 0, stream>>>(gid, cnt, NN);
  zero(xcg, (size_t)GG * HH);
  zero(xsg, (size_t)GG * HH);
  graph_pool_sum<<<cdiv((long long)NHf, BT), BT, 0, stream>>>(B3, gid, xcg, NN);
  graph_pool_sum<<<cdiv((long long)NHf, BT), BT, 0, stream>>>(B4, gid, xsg, NN);
  graph_pool_div<<<cdiv(GG * HH, BT), BT, 0, stream>>>(xcg, cnt, GG);
  graph_pool_div<<<cdiv(GG * HH, BT), BT, 0, stream>>>(xsg, cnt, GG);

  // ---- heads ----
  // causal: bn -> fc1+relu -> bn -> fc2
  bn_small<<<1, HH, 0, stream>>>(xcg, t1, GG, HH);
  mm_small<<<cdiv(GG * DD, BT), BT, 0, stream>>>(t1, cfc1_W, cfc1_b, t2, GG, HH, DD, 1);
  bn_small<<<1, DD, 0, stream>>>(t2, t3, GG, DD);
  mm_small<<<1, GG * CC, 0, stream>>>(t3, cfc2_W, cfc2_b, out_causal, GG, DD, CC, 0);
  // spurious: ... + log_softmax
  bn_small<<<1, HH, 0, stream>>>(xsg, t1, GG, HH);
  mm_small<<<cdiv(GG * DD, BT), BT, 0, stream>>>(t1, sfc1_W, sfc1_b, t2, GG, HH, DD, 1);
  bn_small<<<1, DD, 0, stream>>>(t2, t3, GG, DD);
  mm_small<<<1, GG * CC, 0, stream>>>(t3, sfc2_W, sfc2_b, out_spur, GG, DD, CC, 0);
  log_softmax2<<<1, GG, 0, stream>>>(out_spur, GG);
  // context: concat(xc_g, xs_g[perm]) -> bn -> fc1+relu -> bn -> fc2
  build_xcat<<<cdiv(GG * 256, BT), BT, 0, stream>>>(xcg, xsg, xcat, GG);
  bn_small<<<1, 256, 0, stream>>>(xcat, xcatbn, GG, 256);
  mm_small<<<cdiv(GG * DD, BT), BT, 0, stream>>>(xcatbn, ccat1_W, ccat1_b, t2, GG, 256, DD, 1);
  bn_small<<<1, DD, 0, stream>>>(t2, t3, GG, DD);
  mm_small<<<1, GG * CC, 0, stream>>>(t3, ccat2_W, ccat2_b, out_ctx, GG, DD, CC, 0);
}
